// AttributeHypergraphModel_7713761263856
// MI455X (gfx1250) — matrix operand order repaired and checked by reading, not compiled, not run on hardware
//
#include <hip/hip_runtime.h>

#define NEG_SLOPE 0.2f

typedef __attribute__((ext_vector_type(16))) __bf16 v16bf;
typedef __attribute__((ext_vector_type(8)))  float  v8f;

__device__ __forceinline__ v8f wmma_bf16(v16bf a, v16bf b, v8f c) {
  // D = A(16x32 bf16) x B(32x16 bf16) + C(16x16 f32)
  return __builtin_amdgcn_wmma_f32_16x16x32_bf16(false, a, false, b, (short)0, c,
                                                 false, false);
}

// B-matrix tile: B[k][n] = W[ntile*16 + n][k]  (W row-major [Nout x K], bf16)
// Lane n<16 holds column n, K = {k0..k0+7, k0+16..k0+23} ; lane n>=16 holds column n-16, K shifted by 8
__device__ __forceinline__ v16bf load_b_tile(const __bf16* W, int K, int ntile,
                                             int k0, int lane) {
  int col  = ntile * 16 + (lane & 15);
  int kofs = k0 + ((lane < 16) ? 0 : 8);
  const __bf16* p = W + (size_t)col * K + kofs;
  union { v16bf v; int4 q[2]; } u;
  u.q[0] = *(const int4*)(p);
  u.q[1] = *(const int4*)(p + 16);
  return u.v;
}

// A-matrix tile from bf16 row-major memory.
// Lane l<16: row l, K = {k0..k0+7, k0+16..k0+23}; lane l>=16: row l-16, K shifted by 8.
__device__ __forceinline__ v16bf load_a_bf(const __bf16* rowptr, int k0, int lane) {
  int c1 = k0 + ((lane < 16) ? 0 : 8);
  int c2 = k0 + ((lane < 16) ? 16 : 24);
  union { v16bf v; int4 q[2]; } u;
  u.q[0] = *(const int4*)(rowptr + c1);
  u.q[1] = *(const int4*)(rowptr + c2);
  return u.v;
}

// A-matrix tile gathered from f32 memory with on-the-fly bf16 conversion.
__device__ __forceinline__ v16bf load_a_f32(const float* rowptr, int k0, int lane) {
  const float* p1 = rowptr + k0 + ((lane < 16) ? 0 : 8);
  const float* p2 = rowptr + k0 + ((lane < 16) ? 16 : 24);
  float4 f0 = *(const float4*)(p1);
  float4 f1 = *(const float4*)(p1 + 4);
  float4 f2 = *(const float4*)(p2);
  float4 f3 = *(const float4*)(p2 + 4);
  union { v16bf v; __bf16 e[16]; } a;
  a.e[0]  = (__bf16)f0.x; a.e[1]  = (__bf16)f0.y; a.e[2]  = (__bf16)f0.z; a.e[3]  = (__bf16)f0.w;
  a.e[4]  = (__bf16)f1.x; a.e[5]  = (__bf16)f1.y; a.e[6]  = (__bf16)f1.z; a.e[7]  = (__bf16)f1.w;
  a.e[8]  = (__bf16)f2.x; a.e[9]  = (__bf16)f2.y; a.e[10] = (__bf16)f2.z; a.e[11] = (__bf16)f2.w;
  a.e[12] = (__bf16)f3.x; a.e[13] = (__bf16)f3.y; a.e[14] = (__bf16)f3.z; a.e[15] = (__bf16)f3.w;
  return a.v;
}

__device__ __forceinline__ unsigned f2ord(float f) {
  unsigned u = __float_as_uint(f);
  return (u & 0x80000000u) ? ~u : (u | 0x80000000u);
}
__device__ __forceinline__ float ord2f(unsigned k) {
  unsigned u = (k & 0x80000000u) ? (k & 0x7FFFFFFFu) : ~k;
  return __uint_as_float(u);
}

// ---------------------------------------------------------------------------
// f32 -> bf16 convert
// ---------------------------------------------------------------------------
__global__ void cvt_bf16_kernel(const float* __restrict__ src,
                                __bf16* __restrict__ dst, long n) {
  long i = (long)blockIdx.x * blockDim.x + threadIdx.x;
  if (i < n) dst[i] = (__bf16)src[i];
}

// ---------------------------------------------------------------------------
// Entity embedding: one wave per node. 3 WMMA passes (rp, h-attrs, t-attrs).
// Writes triple_emb in bf16: [h+t (128) | r (128)].
// ---------------------------------------------------------------------------
#define EW 4   // waves per block

__global__ __launch_bounds__(EW * 32)
void entity_kernel(const int* __restrict__ hattr, const int* __restrict__ tattr,
                   const int* __restrict__ ridx,
                   const float* __restrict__ attr_table,
                   const float* __restrict__ rel_table,
                   const __bf16* __restrict__ femb_bf,
                   const float* __restrict__ femb_b,
                   __bf16* __restrict__ Xbf, int N) {
  __shared__ float lds_ap[EW][16][132];   // pitch 132: 132%64==4 -> conflict-free
  __shared__ float lds_rp[EW][128];
  __shared__ float lds_att[EW][16];

  const int wave = threadIdx.x >> 5;
  const int lane = threadIdx.x & 31;
  int n = blockIdx.x * EW + wave;
  if (n >= N) n = N - 1;               // wave-uniform clamp (EXEC stays all-1)

  const float* rbase = rel_table + (size_t)ridx[n] * 128;
  const int m = lane & 15;
  const float* rowH = attr_table + (size_t)hattr[(size_t)n * 16 + m] * 128;
  const float* rowT = attr_table + (size_t)tattr[(size_t)n * 16 + m] * 128;

  v8f acc[8];
  union { v8f v; float e[8]; } uacc;

  // ---- Pass R: all 16 rows = r  ->  rp = row 0 of D ----
  #pragma unroll
  for (int nt = 0; nt < 8; nt++) acc[nt] = (v8f){0,0,0,0,0,0,0,0};
  for (int k0 = 0; k0 < 128; k0 += 32) {
    v16bf a = load_a_f32(rbase, k0, lane);
    #pragma unroll
    for (int nt = 0; nt < 8; nt++)
      acc[nt] = wmma_bf16(a, load_b_tile(femb_bf, 128, nt, k0, lane), acc[nt]);
  }
  if (lane < 16) {
    #pragma unroll
    for (int nt = 0; nt < 8; nt++) {
      uacc.v = acc[nt];
      int de = nt * 16 + lane;
      lds_rp[wave][de] = uacc.e[0] + femb_b[de];   // M=0 row
    }
  }

  float ho[4] = {0,0,0,0}, to[4] = {0,0,0,0};
  const int de0 = lane * 4;

  #pragma unroll 1
  for (int side = 0; side < 2; side++) {
    const float* row = side == 0 ? rowH : rowT;
    #pragma unroll
    for (int nt = 0; nt < 8; nt++) acc[nt] = (v8f){0,0,0,0,0,0,0,0};
    for (int k0 = 0; k0 < 128; k0 += 32) {
      v16bf a = load_a_f32(row, k0, lane);
      #pragma unroll
      for (int nt = 0; nt < 8; nt++)
        acc[nt] = wmma_bf16(a, load_b_tile(femb_bf, 128, nt, k0, lane), acc[nt]);
    }
    // scatter ap (+bias) to LDS
    const int mofs = (lane < 16) ? 0 : 8;
    #pragma unroll
    for (int nt = 0; nt < 8; nt++) {
      uacc.v = acc[nt];
      int col = nt * 16 + m;
      float b = femb_b[col];
      #pragma unroll
      for (int v = 0; v < 8; v++) lds_ap[wave][mofs + v][col] = uacc.e[v] + b;
    }
    __syncthreads();

    // scores + softmax over the 16 attributes (lanes 0..15)
    if (lane < 16) {
      float s = 0.f;
      for (int de = 0; de < 128; de++) s += lds_rp[wave][de] * lds_ap[wave][lane][de];
      float mx = s;
      mx = fmaxf(mx, __shfl_xor(mx, 1));
      mx = fmaxf(mx, __shfl_xor(mx, 2));
      mx = fmaxf(mx, __shfl_xor(mx, 4));
      mx = fmaxf(mx, __shfl_xor(mx, 8));
      float ex = __expf(s - mx);
      float sm = ex;
      sm += __shfl_xor(sm, 1);
      sm += __shfl_xor(sm, 2);
      sm += __shfl_xor(sm, 4);
      sm += __shfl_xor(sm, 8);
      lds_att[wave][lane] = ex / sm;
    }
    __syncthreads();

    // out[de] = sum_a att[a] * ap[a][de]  (each lane owns 4 de's)
    float* o = side == 0 ? ho : to;
    #pragma unroll
    for (int a = 0; a < 16; a++) {
      float w = lds_att[wave][a];
      #pragma unroll
      for (int j = 0; j < 4; j++) o[j] += w * lds_ap[wave][a][de0 + j];
    }
    __syncthreads();  // before next side overwrites lds_ap
  }

  // triple_emb = [h+t | r]  in bf16
  __bf16* xo = Xbf + (size_t)n * 256;
  float4 rv = *(const float4*)(rbase + de0);
  float re[4] = {rv.x, rv.y, rv.z, rv.w};
  #pragma unroll
  for (int j = 0; j < 4; j++) {
    xo[de0 + j]       = (__bf16)(ho[j] + to[j]);
    xo[128 + de0 + j] = (__bf16)re[j];
  }
}

// ---------------------------------------------------------------------------
// Generic bf16 GEMM: C[M x 128] (f32) = A[M x K] (bf16) @ W[128 x K]^T (bf16)
// One wave per 16-row tile; 8 n-tiles x K/32 k-steps of WMMA.
// ---------------------------------------------------------------------------
__global__ __launch_bounds__(256)
void gemm_bf16_kernel(const __bf16* __restrict__ A, const __bf16* __restrict__ W,
                      float* __restrict__ C, int M, int K) {
  const int wave = threadIdx.x >> 5;
  const int lane = threadIdx.x & 31;
  const int mt = blockIdx.x * 8 + wave;
  const int m0 = mt * 16;
  if (m0 >= M) return;                       // wave-uniform

  int rowA = m0 + (lane & 15);
  if (rowA >= M) rowA = M - 1;
  const __bf16* abase = A + (size_t)rowA * K;

  v8f acc[8];
  #pragma unroll
  for (int nt = 0; nt < 8; nt++) acc[nt] = (v8f){0,0,0,0,0,0,0,0};

  for (int k0 = 0; k0 < K; k0 += 32) {
    v16bf a = load_a_bf(abase, k0, lane);
    #pragma unroll
    for (int nt = 0; nt < 8; nt++)
      acc[nt] = wmma_bf16(a, load_b_tile(W, K, nt, k0, lane), acc[nt]);
  }

  const int coll = lane & 15;
  const int mofs = (lane < 16) ? 0 : 8;
  union { v8f v; float e[8]; } u;
  #pragma unroll
  for (int nt = 0; nt < 8; nt++) {
    u.v = acc[nt];
    int col = nt * 16 + coll;
    #pragma unroll
    for (int v = 0; v < 8; v++) {
      int row = m0 + mofs + v;
      if (row < M) C[(size_t)row * 128 + col] = u.e[v];
    }
  }
}

// ---------------------------------------------------------------------------
// alpha_s/alpha_d: one wave per node; dot(H[n], a_src/a_dst)
// ---------------------------------------------------------------------------
__global__ __launch_bounds__(256)
void alpha_kernel(const float* __restrict__ H, const float* __restrict__ asrc,
                  const float* __restrict__ adst, float* __restrict__ as_,
                  float* __restrict__ ad_, int N) {
  const int wave = threadIdx.x >> 5;
  const int lane = threadIdx.x & 31;
  const int n = blockIdx.x * 8 + wave;
  if (n >= N) return;
  const float* h = H + (size_t)n * 128 + lane * 4;
  float s = 0.f, d = 0.f;
  #pragma unroll
  for (int j = 0; j < 4; j++) {
    float v = h[j];
    s += v * asrc[lane * 4 + j];
    d += v * adst[lane * 4 + j];
  }
  #pragma unroll
  for (int m = 16; m >= 1; m >>= 1) {
    s += __shfl_xor(s, m);
    d += __shfl_xor(d, m);
  }
  if (lane == 0) { as_[n] = s; ad_[n] = d; }
}

// ---------------------------------------------------------------------------
// Per-layer init: out = bias (broadcast), mkey = ord(-inf), denom = 0
// ---------------------------------------------------------------------------
__global__ void init_kernel(unsigned* __restrict__ mkey, float* __restrict__ denom,
                            float* __restrict__ out, const float* __restrict__ bias,
                            int N) {
  long t = (long)blockIdx.x * blockDim.x + threadIdx.x;
  if (t >= (long)N * 128) return;
  int d = (int)(t & 127);
  out[t] = bias[d];
  if (d == 0) {
    mkey[t >> 7]  = f2ord(-__builtin_inff());
    denom[t >> 7] = 0.f;
  }
}

__device__ __forceinline__ void edge_sd(const int* ei, int E, int e, int& src, int& dst) {
  if (e < E) { src = ei[e]; dst = ei[E + e]; }
  else       { src = dst = e - E; }
}

__global__ void edge_max_kernel(const int* __restrict__ ei, int E, int N,
                                const float* __restrict__ as_,
                                const float* __restrict__ ad_,
                                unsigned* __restrict__ mkey) {
  int e = blockIdx.x * blockDim.x + threadIdx.x;
  if (e >= E + N) return;
  int src, dst; edge_sd(ei, E, e, src, dst);
  float v = as_[src] + ad_[dst];
  v = v > 0.f ? v : NEG_SLOPE * v;
  atomicMax(mkey + dst, f2ord(v));
}

__global__ void edge_denom_kernel(const int* __restrict__ ei, int E, int N,
                                  const float* __restrict__ as_,
                                  const float* __restrict__ ad_,
                                  const unsigned* __restrict__ mkey,
                                  float* __restrict__ denom) {
  int e = blockIdx.x * blockDim.x + threadIdx.x;
  if (e >= E + N) return;
  int src, dst; edge_sd(ei, E, e, src, dst);
  float v = as_[src] + ad_[dst];
  v = v > 0.f ? v : NEG_SLOPE * v;
  atomicAdd(denom + dst, __expf(v - ord2f(mkey[dst])));
}

// one wave per edge: out[dst] += alpha * H[src]
__global__ __launch_bounds__(256)
void edge_agg_kernel(const int* __restrict__ ei, int E, int N,
                     const float* __restrict__ as_, const float* __restrict__ ad_,
                     const unsigned* __restrict__ mkey,
                     const float* __restrict__ denom,
                     const float* __restrict__ H, float* __restrict__ out) {
  const int wave = threadIdx.x >> 5;
  const int lane = threadIdx.x & 31;
  const int e = blockIdx.x * 8 + wave;
  if (e >= E + N) return;
  int src, dst; edge_sd(ei, E, e, src, dst);
  float v = as_[src] + ad_[dst];
  v = v > 0.f ? v : NEG_SLOPE * v;
  float w = __expf(v - ord2f(mkey[dst])) / denom[dst];
  float4 hv = *(const float4*)(H + (size_t)src * 128 + lane * 4);
  float* o = out + (size_t)dst * 128 + lane * 4;
  atomicAdd(o + 0, w * hv.x);
  atomicAdd(o + 1, w * hv.y);
  atomicAdd(o + 2, w * hv.z);
  atomicAdd(o + 3, w * hv.w);
}

// ---------------------------------------------------------------------------
extern "C" void kernel_launch(void* const* d_in, const int* in_sizes, int n_in,
                              void* d_out, int out_size, void* d_ws, size_t ws_size,
                              hipStream_t stream) {
  const int*   hattr      = (const int*)d_in[0];
  const int*   tattr      = (const int*)d_in[1];
  const int*   ridx       = (const int*)d_in[2];
  const int*   edge_index = (const int*)d_in[3];
  const float* attr_table = (const float*)d_in[4];
  const float* rel_table  = (const float*)d_in[5];
  const float* femb_w     = (const float*)d_in[6];
  const float* femb_b     = (const float*)d_in[7];
  const float* g1_w       = (const float*)d_in[8];
  const float* g1_asrc    = (const float*)d_in[9];
  const float* g1_adst    = (const float*)d_in[10];
  const float* g1_b       = (const float*)d_in[11];
  const float* g2_w       = (const float*)d_in[12];
  const float* g2_asrc    = (const float*)d_in[13];
  const float* g2_adst    = (const float*)d_in[14];
  const float* g2_b       = (const float*)d_in[15];

  const int N  = in_sizes[2];
  const int E  = in_sizes[3] / 2;
  const int Et = E + N;
  float* out2 = (float*)d_out;

  // workspace carve-out (256B aligned)
  char* ws = (char*)d_ws;
  size_t o = 0;
  auto take = [&](size_t bytes) -> char* {
    char* p = ws + o;
    o = (o + bytes + 255) & ~(size_t)255;
    return p;
  };
  __bf16*   femb_bf = (__bf16*)take((size_t)128 * 128 * 2);
  __bf16*   g1_bf   = (__bf16*)take((size_t)128 * 256 * 2);
  __bf16*   g2_bf   = (__bf16*)take((size_t)128 * 128 * 2);
  __bf16*   Xbf     = (__bf16*)take((size_t)N * 256 * 2);
  float*    Hf      = (float*)take((size_t)N * 128 * 4);
  float*    O1      = (float*)take((size_t)N * 128 * 4);
  __bf16*   X2      = (__bf16*)take((size_t)N * 128 * 2);
  float*    as_     = (float*)take((size_t)N * 4);
  float*    ad_     = (float*)take((size_t)N * 4);
  unsigned* mkey    = (unsigned*)take((size_t)N * 4);
  float*    denom   = (float*)take((size_t)N * 4);

  // 1) weights -> bf16
  cvt_bf16_kernel<<<(128 * 128 + 255) / 256, 256, 0, stream>>>(femb_w, femb_bf, 128L * 128);
  cvt_bf16_kernel<<<(128 * 256 + 255) / 256, 256, 0, stream>>>(g1_w, g1_bf, 128L * 256);
  cvt_bf16_kernel<<<(128 * 128 + 255) / 256, 256, 0, stream>>>(g2_w, g2_bf, 128L * 128);

  // 2) entity embeddings + triple concat (bf16)
  entity_kernel<<<(N + EW - 1) / EW, EW * 32, 0, stream>>>(
      hattr, tattr, ridx, attr_table, rel_table, femb_bf, femb_b, Xbf, N);

  const int mtiles = (N + 15) / 16;
  const int gemm_blocks = (mtiles + 7) / 8;
  const int nodew_blocks = (N + 7) / 8;
  const int edge_blocks = (Et + 255) / 256;
  const int eagg_blocks = (Et + 7) / 8;
  const int init_blocks = (int)(((long)N * 128 + 255) / 256);

  // ---- GAT layer 1 ----
  gemm_bf16_kernel<<<gemm_blocks, 256, 0, stream>>>(Xbf, g1_bf, Hf, N, 256);
  alpha_kernel<<<nodew_blocks, 256, 0, stream>>>(Hf, g1_asrc, g1_adst, as_, ad_, N);
  init_kernel<<<init_blocks, 256, 0, stream>>>(mkey, denom, O1, g1_b, N);
  edge_max_kernel<<<edge_blocks, 256, 0, stream>>>(edge_index, E, N, as_, ad_, mkey);
  edge_denom_kernel<<<edge_blocks, 256, 0, stream>>>(edge_index, E, N, as_, ad_, mkey, denom);
  edge_agg_kernel<<<eagg_blocks, 256, 0, stream>>>(edge_index, E, N, as_, ad_, mkey, denom, Hf, O1);

  // ---- GAT layer 2 ----
  cvt_bf16_kernel<<<(int)(((long)N * 128 + 255) / 256), 256, 0, stream>>>(O1, X2, (long)N * 128);
  gemm_bf16_kernel<<<gemm_blocks, 256, 0, stream>>>(X2, g2_bf, Hf, N, 128);
  alpha_kernel<<<nodew_blocks, 256, 0, stream>>>(Hf, g2_asrc, g2_adst, as_, ad_, N);
  init_kernel<<<init_blocks, 256, 0, stream>>>(mkey, denom, out2, g2_b, N);
  edge_max_kernel<<<edge_blocks, 256, 0, stream>>>(edge_index, E, N, as_, ad_, mkey);
  edge_denom_kernel<<<edge_blocks, 256, 0, stream>>>(edge_index, E, N, as_, ad_, mkey, denom);
  edge_agg_kernel<<<eagg_blocks, 256, 0, stream>>>(edge_index, E, N, as_, ad_, mkey, denom, Hf, out2);
}
